// Head_37331855737481
// MI455X (gfx1250) — matrix operand order repaired and checked by reading, not compiled, not run on hardware
//
#include <hip/hip_runtime.h>

// ---------------------------------------------------------------------------
// Single-head causal attention, B=2048, T=128, C=384, HS=64.
// HBM-bound (~470MB moved, ~20us floor at 23.3 TB/s): read x once (non-
// temporal), bf16 WMMA for all GEMMs with fp32 accumulation, all intermediates
// in LDS/registers. Phase-2 is register-blocked over weight column slabs so
// each block reads each weight fragment at most 2x (vs 8x naive), keeping L2
// traffic negligible next to the 403MB x stream.
// ---------------------------------------------------------------------------

typedef __attribute__((ext_vector_type(16))) __bf16       v16bf;
typedef __attribute__((ext_vector_type(8)))  float        v8f;
typedef __attribute__((ext_vector_type(4)))  float        f32x4;
typedef __attribute__((ext_vector_type(4)))  unsigned int u32x4;

union FragBF { v16bf v; u32x4 q[2]; };

#define Tn   128
#define Cn   384
#define HSn  64
#define XS   392   // xs row stride (bf16 elems), padded vs 384 to break bank alignment
#define QS   72    // q/k row stride, padded vs 64
#define PS   136   // probs / v^T row stride, padded vs 128

__device__ __forceinline__ unsigned short f2bf(float f) {
  unsigned int u = __float_as_uint(f);
  u += 0x7FFFu + ((u >> 16) & 1u);           // round-to-nearest-even
  return (unsigned short)(u >> 16);
}

__device__ __forceinline__ v8f wmma_bf16(v16bf a, v16bf b, v8f c) {
  // (neg_a, A, neg_b, B, c_mod, C, reuse_a, reuse_b)
  return __builtin_amdgcn_wmma_f32_16x16x32_bf16(false, a, false, b, (short)0, c,
                                                 false, false);
}

// Prep: convert Wq/Wk/Wv [C][HS] fp32 -> bf16 transposed Wt[3][HS][C] in d_ws,
// so WMMA B-fragments (N=row, K contiguous) are two contiguous b128 loads.
__global__ void wt_prep(const float* __restrict__ Wq, const float* __restrict__ Wk,
                        const float* __restrict__ Wv, unsigned short* __restrict__ wt) {
  int idx = blockIdx.x * blockDim.x + threadIdx.x;
  if (idx >= 3 * Cn * HSn) return;
  int mat = idx / (Cn * HSn);
  int rem = idx - mat * (Cn * HSn);
  int c = rem / HSn, h = rem - c * HSn;
  const float* W = (mat == 0) ? Wq : (mat == 1) ? Wk : Wv;
  wt[mat * (HSn * Cn) + h * Cn + c] = f2bf(W[rem]);
}

__global__ __launch_bounds__(256, 1) void attn_head(
    const float* __restrict__ x, const unsigned short* __restrict__ wt,
    float* __restrict__ out) {
  // Region0: x staged bf16 (phase 1-2), then reused for probs (phase 3-4).
  __shared__ __align__(16) unsigned short sm0[Tn * XS];      // 100352 B
  __shared__ __align__(16) unsigned short qsm[Tn * QS];      //  18432 B
  __shared__ __align__(16) unsigned short ksm[Tn * QS];      //  18432 B
  __shared__ __align__(16) unsigned short vtm[HSn * PS];     //  17408 B (v^T)

  const int tid  = threadIdx.x;
  const int b    = blockIdx.x;
  const int wave = tid >> 5;
  const int lane = tid & 31;
  const int g    = lane >> 4;   // half-wave group
  const int ln   = lane & 15;   // M (A-frag) or N (B-frag / C-frag)

  // -------- Phase 1: stage x[b] fp32 -> bf16 into LDS (single HBM read) ----
  const f32x4* xv = (const f32x4*)(x + (size_t)b * (Tn * Cn));
  for (int i = tid; i < (Tn * Cn) / 4; i += 256) {
    f32x4 f = __builtin_nontemporal_load(xv + i);
    int e0 = i * 4;
    int row = e0 / Cn, col = e0 - row * Cn;
    unsigned int lo = (unsigned)f2bf(f[0]) | ((unsigned)f2bf(f[1]) << 16);
    unsigned int hi = (unsigned)f2bf(f[2]) | ((unsigned)f2bf(f[3]) << 16);
    unsigned int* dst = (unsigned int*)&sm0[row * XS + col];
    dst[0] = lo;
    dst[1] = hi;
  }
  __syncthreads();

  // -------- Phase 2: q,k,v = x @ W, register-blocked over weight slabs -----
  // 24 units = (mat 0..2) x (n0 0..3) x (m-half 0..1); 3 units per wave.
  // B fragments for the whole K=384 slab are held in 96 VGPRs and reused
  // across 4 M-tiles -> per-block L2 weight traffic ~2x the unique 147KB.
  for (int u = wave; u < 24; u += 8) {
    const int mat   = u >> 3;                    // 0=q 1=k 2=v
    const int n0    = ((u >> 1) & 3) << 4;
    const int mhalf = u & 1;
    const unsigned short* brow = wt + mat * (HSn * Cn) + (n0 + ln) * Cn;
    FragBF bfr[12];
#pragma unroll
    for (int ks = 0; ks < 12; ++ks) {            // B: K=8g+e / K=16+8g+(e-8)
      bfr[ks].q[0] = *(const u32x4*)(brow + ks * 32 + 8 * g);
      bfr[ks].q[1] = *(const u32x4*)(brow + ks * 32 + 16 + 8 * g);
    }
#pragma unroll
    for (int mt = 0; mt < 4; ++mt) {
      const int m0 = (mhalf * 4 + mt) << 4;
      const unsigned short* arow = sm0 + (m0 + ln) * XS;
      v8f acc = {};
#pragma unroll
      for (int ks = 0; ks < 12; ++ks) {
        FragBF a;                                // A: K=8g+e / K=16+8g+(e-8)
        a.q[0] = *(const u32x4*)(arow + ks * 32 + 8 * g);
        a.q[1] = *(const u32x4*)(arow + ks * 32 + 16 + 8 * g);
        acc = wmma_bf16(a.v, bfr[ks].v, acc);
      }
      if (mat == 2) {                            // store v transposed: vtm[hs][t]
#pragma unroll
        for (int r = 0; r < 8; ++r)
          vtm[(n0 + ln) * PS + m0 + 8 * g + r] = f2bf(acc[r]);
      } else {
        unsigned short* dst = (mat == 0) ? qsm : ksm;
        const float scale = (mat == 0) ? 0.05103103630798287f : 1.0f;  // C^-0.5
#pragma unroll
        for (int r = 0; r < 8; ++r)
          dst[(m0 + 8 * g + r) * QS + n0 + ln] = f2bf(acc[r] * scale);
      }
    }
  }
  __syncthreads();

  // -------- Phase 3: scores = q k^T, causal mask, softmax ------------------
  const int i0 = wave * 16;                      // each wave owns 16 rows
  v8f sc[8];
  {
    const unsigned short* qrow = qsm + (i0 + ln) * QS;
#pragma unroll
    for (int jt = 0; jt < 8; ++jt) {
      const unsigned short* krow = ksm + (jt * 16 + ln) * QS;
      v8f acc = {};
#pragma unroll
      for (int kc = 0; kc < HSn; kc += 32) {
        FragBF a, bm;
        a.q[0]  = *(const u32x4*)(qrow + kc + 8 * g);
        a.q[1]  = *(const u32x4*)(qrow + kc + 16 + 8 * g);
        bm.q[0] = *(const u32x4*)(krow + kc + 8 * g);
        bm.q[1] = *(const u32x4*)(krow + kc + 16 + 8 * g);
        acc = wmma_bf16(a.v, bm.v, acc);
      }
      sc[jt] = acc;
    }
  }
  // C/D layout: element r of lane (g,ln) is (M = i0+8g+r, N = jt*16+ln).
#pragma unroll
  for (int r = 0; r < 8; ++r) {
    const int row = i0 + 8 * g + r;
    float m = -INFINITY;
#pragma unroll
    for (int jt = 0; jt < 8; ++jt) {
      const int col = jt * 16 + ln;
      float s = (col > row) ? -INFINITY : sc[jt][r];
      sc[jt][r] = s;
      m = fmaxf(m, s);
    }
    m = fmaxf(m, __shfl_xor(m, 1));              // 16-lane row reduction
    m = fmaxf(m, __shfl_xor(m, 2));              // (masks <=8 stay in half-wave)
    m = fmaxf(m, __shfl_xor(m, 4));
    m = fmaxf(m, __shfl_xor(m, 8));
    float sum = 0.0f;
#pragma unroll
    for (int jt = 0; jt < 8; ++jt) {
      float p = __expf(sc[jt][r] - m);
      sc[jt][r] = p;
      sum += p;
    }
    sum += __shfl_xor(sum, 1);
    sum += __shfl_xor(sum, 2);
    sum += __shfl_xor(sum, 4);
    sum += __shfl_xor(sum, 8);
    const float inv = 1.0f / sum;
#pragma unroll
    for (int jt = 0; jt < 8; ++jt) sc[jt][r] *= inv;
  }
  // Write probs bf16 into region0 (x staging is dead past the last barrier).
  unsigned short* ps = sm0;
#pragma unroll
  for (int jt = 0; jt < 8; ++jt) {
#pragma unroll
    for (int r = 0; r < 8; ++r)
      ps[(i0 + 8 * g + r) * PS + jt * 16 + ln] = f2bf(sc[jt][r]);
  }
  __syncthreads();

  // -------- Phase 4: out = P @ v  (32 tiles, 4 K-steps) --------------------
  for (int t = wave; t < 32; t += 8) {           // uniform: 4 tiles per wave
    const int m0 = (t >> 2) << 4;
    const int n0 = (t & 3) << 4;
    const unsigned short* arow = ps + (m0 + ln) * PS;
    const unsigned short* brow = vtm + (n0 + ln) * PS;   // v^T rows contiguous in K
    v8f acc = {};
#pragma unroll
    for (int kc = 0; kc < Tn; kc += 32) {
      FragBF a, bm;
      a.q[0]  = *(const u32x4*)(arow + kc + 8 * g);
      a.q[1]  = *(const u32x4*)(arow + kc + 16 + 8 * g);
      bm.q[0] = *(const u32x4*)(brow + kc + 8 * g);
      bm.q[1] = *(const u32x4*)(brow + kc + 16 + 8 * g);
      acc = wmma_bf16(a.v, bm.v, acc);
    }
#pragma unroll
    for (int r = 0; r < 8; ++r) {
      float* dst = out + ((size_t)b * Tn + m0 + 8 * g + r) * HSn + n0 + ln;
      __builtin_nontemporal_store(acc[r], dst);
    }
  }
}

extern "C" void kernel_launch(void* const* d_in, const int* in_sizes, int n_in,
                              void* d_out, int out_size, void* d_ws, size_t ws_size,
                              hipStream_t stream) {
  const float* x  = (const float*)d_in[0];
  const float* Wq = (const float*)d_in[1];
  const float* Wk = (const float*)d_in[2];
  const float* Wv = (const float*)d_in[3];
  float* out = (float*)d_out;
  unsigned short* wt = (unsigned short*)d_ws;   // 3*64*384*2 = 147456 B of scratch

  wt_prep<<<(3 * Cn * HSn + 255) / 256, 256, 0, stream>>>(Wq, Wk, Wv, wt);
  attn_head<<<2048, 256, 0, stream>>>(x, wt, out);
}
